// Model_79044578115797
// MI455X (gfx1250) — compile-verified, matrix operand order checked by reading, not run on hardware
//
#include <hip/hip_runtime.h>
#include <hip/hip_fp16.h>

// ---------------- problem constants ----------------
constexpr int Bb = 4, Rr = 16, Cc = 511, Nn = 512, Hh = 64, H2 = 128;
constexpr int T1 = 5, T2 = 5;
constexpr float NEGV = 1e10f;
constexpr float LOG2E = 1.4426950408889634f;

typedef __attribute__((ext_vector_type(16))) _Float16 v16h;
typedef __attribute__((ext_vector_type(8)))  _Float16 h8;
typedef __attribute__((ext_vector_type(8)))  float    v8f;

// ---------------- fast math helpers ----------------
__device__ __forceinline__ float fast_exp2(float x) {
    return __builtin_amdgcn_exp2f(x);
}
__device__ __forceinline__ float fast_tanh(float x) {
#if __has_builtin(__builtin_amdgcn_tanhf)
    return __builtin_amdgcn_tanhf(x);
#else
    // tanh(x) = 1 - 2/(e^{2x}+1), e^{2x} = exp2(2*log2(e)*x)
    float t = __builtin_amdgcn_exp2f(x * 2.8853900817779268f);
    return 1.0f - 2.0f * __builtin_amdgcn_rcpf(t + 1.0f);
#endif
}

// ---------------- WMMA fragment helper (f16 16x16x32, ISA 7.12.2 layout) -----
// Per-lane halves map to two contiguous 8-half runs of the lane's row:
//   halves 0..7  <- row[kb + 8*hi .. +8)        (K = kb + i + 8*hi)
//   halves 8..15 <- row[kb + 16 + 8*hi .. +8)   (K = kb + i + 8 + 8*hi)
// A: rowbase = &mat[m][0] (row-major MxK). B: rowbase = &matT[n][0] (N-major NxK,
// i.e. the weight matrix stored transposed so columns are contiguous).
__device__ __forceinline__ v16h load_frag(const _Float16* rowbase, int kb, int hi) {
    h8 lo = *(const h8*)(rowbase + kb + 8 * hi);
    h8 hh = *(const h8*)(rowbase + kb + 16 + 8 * hi);
    return __builtin_shufflevector(lo, hh, 0, 1, 2, 3, 4, 5, 6, 7,
                                   8, 9, 10, 11, 12, 13, 14, 15);
}
__device__ __forceinline__ v8f wmma16(v16h a, v16h b, v8f c) {
    return __builtin_amdgcn_wmma_f32_16x16x32_f16(false, a, false, b,
                                                  (short)0, c, false, false);
}

// ---------------- small prep kernels ----------------
// convert f32 weight [K][Nc] -> f16 transposed [Nc][K] (for contiguous B frags)
__global__ void k_cvt_h_t(const float* __restrict__ s, _Float16* __restrict__ d,
                          int K, int Nc) {
    int i = blockIdx.x * 256 + threadIdx.x;
    if (i < K * Nc) {
        int k = i / Nc, n = i % Nc;
        d[n * K + k] = (_Float16)s[i];
    }
}

__global__ __launch_bounds__(128)
void k_xa_terms(const float* __restrict__ x_a, const float* __restrict__ ap,
                const float* __restrict__ act, const float* __restrict__ x_b,
                const float* __restrict__ Wx1a, const float* __restrict__ bx1a,
                const float* __restrict__ Wx1b, const float* __restrict__ bx1b,
                float* __restrict__ xa_t, float* __restrict__ xb_t) {
    int id = blockIdx.x * 128 + threadIdx.x;
    if (id >= Bb * Nn) return;
    int b = id / Nn, n = id % Nn;
    float m = -3.4e38f;
    for (int r = 0; r < Rr; ++r) {
        size_t o = ((size_t)(b * Rr + r)) * Nn + n;
        m = fmaxf(m, x_a[o] * (ap[o] + act[o]));
    }
    float xb = x_b[id];
    for (int h = 0; h < Hh; ++h) {
        xa_t[(size_t)id * Hh + h] = fmaf(m, Wx1a[h], bx1a[h]);
        xb_t[(size_t)id * Hh + h] = fmaf(xb, Wx1b[h], bx1b[h]);
    }
}

// presence MLP: h2[b,c,n] = relu(relu(e@W1p+b1p)@W2p+b2p)/TAU ; also d_t[b,n,c]=edge[b,c,n,0]
__global__ __launch_bounds__(256)
void k_presence(const float* __restrict__ edge,
                const float* __restrict__ W1p, const float* __restrict__ b1p,
                const float* __restrict__ W2p, const float* __restrict__ b2p,
                float* __restrict__ h2buf, float* __restrict__ d_t) {
    __shared__ float w1s[192], b1s[64], w2s[64];
    __shared__ float b2s;
    int t = threadIdx.x;
    if (t < 192) w1s[t] = W1p[t];
    if (t < 64) { b1s[t] = b1p[t]; w2s[t] = W2p[t]; }
    if (t == 0) b2s = b2p[0];
    __syncthreads();
    long id = (long)blockIdx.x * 256 + t;
    if (id >= (long)Bb * Cc * Nn) return;
    int n = (int)(id % Nn);
    long bc = id / Nn;
    int c = (int)(bc % Cc);
    int b = (int)(bc / Cc);
    const float* e = edge + (size_t)id * 3;
    float e0 = e[0], e1 = e[1], e2 = e[2];
    float acc = 0.f;
    for (int j = 0; j < 64; ++j) {
        float h1 = fmaf(e0, w1s[j], fmaf(e1, w1s[64 + j], fmaf(e2, w1s[128 + j], b1s[j])));
        h1 = h1 > 0.f ? h1 : 0.f;
        acc = fmaf(h1, w2s[j], acc);
    }
    acc += b2s;
    acc = acc > 0.f ? acc : 0.f;
    acc *= 0.5f;  // / TAU (=2)
    h2buf[id] = acc;
    d_t[((size_t)b * Nn + n) * Cc + c] = e0;
}

// softmax over n for each (b,c); write transposed p_t[b][n][c]
__global__ __launch_bounds__(256)
void k_softmax(const float* __restrict__ h2buf, const float* __restrict__ avail,
               float* __restrict__ p_t) {
    __shared__ float red[256];
    int t = threadIdx.x;
    int c = blockIdx.x % Cc;
    int b = blockIdx.x / Cc;
    const float* row = h2buf + (size_t)(b * Cc + c) * Nn;
    const float* av = avail + (size_t)b * Nn;
    float l[2];
#pragma unroll
    for (int i = 0; i < 2; ++i) {
        int n = t + i * 256;
        float mask = (n == c ? 0.f : 1.f) * av[n];
        l[i] = row[n] * mask - (1.f - mask) * NEGV;
    }
    red[t] = fmaxf(l[0], l[1]);
    __syncthreads();
    for (int s = 128; s > 0; s >>= 1) {
        if (t < s) red[t] = fmaxf(red[t], red[t + s]);
        __syncthreads();
    }
    float mx = red[0];
    __syncthreads();
    float e0 = fast_exp2((l[0] - mx) * LOG2E);
    float e1 = fast_exp2((l[1] - mx) * LOG2E);
    red[t] = e0 + e1;
    __syncthreads();
    for (int s = 128; s > 0; s >>= 1) {
        if (t < s) red[t] += red[t + s];
        __syncthreads();
    }
    float inv = 1.f / red[0];
    p_t[((size_t)b * Nn + t) * Cc + c] = e0 * inv;
    p_t[((size_t)b * Nn + t + 256) * Cc + c] = e1 * inv;
}

// ------------- T1 iteration: dual-channel einsum + fused WMMA GEMM -------------
__global__ __launch_bounds__(128)
void k_iter1(const float* __restrict__ p_t, const float* __restrict__ d_t,
             const float* __restrict__ ua_in, const float* __restrict__ ub_in,
             float* __restrict__ ua_out, float* __restrict__ ub_out,
             const _Float16* __restrict__ wl1a_t, const _Float16* __restrict__ wl1b_t,
             const float* __restrict__ We1a, const float* __restrict__ be1a,
             const float* __restrict__ We1b, const float* __restrict__ be1b,
             const float* __restrict__ bl1a, const float* __restrict__ bl1b,
             const float* __restrict__ xa_t, const float* __restrict__ xb_t) {
    constexpr int CHUNK = 64;
    __shared__ float u_sh[2][CHUNK][64];
    __shared__ float p_sh[16][CHUNK];
    __shared__ float d_sh[16][CHUNK];
    __shared__ __align__(16) _Float16 l_sh[2][16][64];
    __shared__ __align__(16) _Float16 w_sh[2][64][64];  // [ch][N][K] (transposed)

    int t = threadIdx.x;
    int b = blockIdx.x >> 5;            // N/16 = 32 tiles
    int n0 = (blockIdx.x & 31) * 16;

    _Float16* w0 = &w_sh[0][0][0];
    _Float16* w1 = &w_sh[1][0][0];
    for (int idx = t; idx < 64 * 64; idx += 128) {
        w0[idx] = wl1a_t[idx];
        w1[idx] = wl1b_t[idx];
    }

    int h = t & 63, ns = t >> 6;        // thread: one h, 8 rows (r = ns + 2j)
    float wa = We1a[h], ba = be1a[h], wb = We1b[h], bbv = be1b[h];
    float acc_a[8], acc_b[8];
#pragma unroll
    for (int j = 0; j < 8; ++j) { acc_a[j] = 0.f; acc_b[j] = 0.f; }

    for (int c0 = 0; c0 < Cc; c0 += CHUNK) {
        int cn = (Cc - c0 < CHUNK) ? (Cc - c0) : CHUNK;
        __syncthreads();
        for (int idx = t; idx < cn * 64; idx += 128) {
            int cc = idx >> 6, hh = idx & 63;
            u_sh[0][cc][hh] = ua_in[((size_t)b * Nn + c0 + cc) * Hh + hh];
            u_sh[1][cc][hh] = ub_in[((size_t)b * Nn + c0 + cc) * Hh + hh];
        }
        for (int idx = t; idx < 16 * cn; idx += 128) {
            int r = idx & 15, cc = idx >> 4;
            size_t o = ((size_t)b * Nn + n0 + r) * Cc + c0 + cc;
            p_sh[r][cc] = p_t[o];
            d_sh[r][cc] = d_t[o];
        }
        // prefetch next chunk of p/d rows (gfx1250 global_prefetch_b8)
        if (t < 16 && c0 + CHUNK < Cc) {
            size_t o = ((size_t)b * Nn + n0 + t) * Cc + c0 + CHUNK;
            __builtin_prefetch(&p_t[o], 0, 0);
            __builtin_prefetch(&d_t[o], 0, 0);
        }
        __syncthreads();
        for (int cc = 0; cc < cn; ++cc) {
            float ua = u_sh[0][cc][h];
            float ub = u_sh[1][cc][h];
#pragma unroll
            for (int j = 0; j < 8; ++j) {
                int r = ns + (j << 1);
                float dv = d_sh[r][cc], pv = p_sh[r][cc];
                acc_a[j] = fmaf(pv * fast_tanh(fmaf(dv, wa, ba)), ua, acc_a[j]);
                acc_b[j] = fmaf(pv * fast_tanh(fmaf(dv, wb, bbv)), ub, acc_b[j]);
            }
        }
    }
#pragma unroll
    for (int j = 0; j < 8; ++j) {
        int r = ns + (j << 1);
        l_sh[0][r][h] = (_Float16)acc_a[j];
        l_sh[1][r][h] = (_Float16)acc_b[j];
    }
    __syncthreads();

    // fused GEMM: u_new = relu(l @ Wl1 + bl1 + x_term), via v_wmma_f32_16x16x32_f16
    // wave -> (channel, 2 column tiles); A fragments shared across the tiles.
    int wv = t >> 5, lane = t & 31, m = lane & 15, hi = lane >> 4;
    int ch = wv >> 1;
    int ct0 = (wv & 1) * 2;
    v16h A0 = load_frag(&l_sh[ch][m][0], 0, hi);
    v16h A1 = load_frag(&l_sh[ch][m][0], 32, hi);
    const float* bias = ch ? bl1b : bl1a;
    const float* xt = ch ? xb_t : xa_t;
    float* uo = ch ? ub_out : ua_out;
    for (int ct = ct0; ct < ct0 + 2; ++ct) {   // wave-uniform: EXEC stays full
        v8f acc = {0.f, 0.f, 0.f, 0.f, 0.f, 0.f, 0.f, 0.f};
        acc = wmma16(A0, load_frag(&w_sh[ch][ct * 16 + m][0], 0, hi), acc);
        acc = wmma16(A1, load_frag(&w_sh[ch][ct * 16 + m][0], 32, hi), acc);
#pragma unroll
        for (int v = 0; v < 8; ++v) {
            int row = v + hi * 8, hc = ct * 16 + m, ng = n0 + row;
            float val = acc[v] + bias[hc] + xt[((size_t)b * Nn + ng) * Hh + hc];
            uo[((size_t)b * Nn + ng) * Hh + hc] = val > 0.f ? val : 0.f;
        }
    }
}

// x2_term = u_cat @ Wx2 + bx2  (pure WMMA GEMM, no relu)
__global__ __launch_bounds__(128)
void k_x2term(const float* __restrict__ ua, const float* __restrict__ ub,
              const _Float16* __restrict__ wx2_t, const float* __restrict__ bx2,
              float* __restrict__ x2t) {
    __shared__ __align__(16) _Float16 a_sh[16][128];
    __shared__ __align__(16) _Float16 w_sh[128][128];   // [N][K] transposed
    int t = threadIdx.x;
    int b = blockIdx.x >> 5;
    int n0 = (blockIdx.x & 31) * 16;
    _Float16* wdst = &w_sh[0][0];
    for (int idx = t; idx < 128 * 128; idx += 128) wdst[idx] = wx2_t[idx];
    for (int idx = t; idx < 16 * 128; idx += 128) {
        int r = idx >> 7, hh = idx & 127;
        float v = (hh < 64) ? ua[((size_t)b * Nn + n0 + r) * Hh + hh]
                            : ub[((size_t)b * Nn + n0 + r) * Hh + hh - 64];
        a_sh[r][hh] = (_Float16)v;
    }
    __syncthreads();
    int wv = t >> 5, lane = t & 31, m = lane & 15, hi = lane >> 4;
    v16h A[4];
#pragma unroll
    for (int kk = 0; kk < 4; ++kk) A[kk] = load_frag(&a_sh[m][0], kk * 32, hi);
    for (int ct = wv * 2; ct < wv * 2 + 2; ++ct) {
        v8f acc = {0.f, 0.f, 0.f, 0.f, 0.f, 0.f, 0.f, 0.f};
#pragma unroll
        for (int kk = 0; kk < 4; ++kk)
            acc = wmma16(A[kk], load_frag(&w_sh[ct * 16 + m][0], kk * 32, hi), acc);
#pragma unroll
        for (int v = 0; v < 8; ++v) {
            int row = v + hi * 8, hc = ct * 16 + m, ng = n0 + row;
            x2t[((size_t)b * Nn + ng) * H2 + hc] = acc[v] + bx2[hc];
        }
    }
}

// ------------- T2 iteration: 2H einsum + fused WMMA GEMM -------------
__global__ __launch_bounds__(128)
void k_iter2(const float* __restrict__ p_t, const float* __restrict__ d_t,
             const float* __restrict__ g_in, float* __restrict__ g_out,
             const _Float16* __restrict__ wl2_t,
             const float* __restrict__ We2, const float* __restrict__ be2,
             const float* __restrict__ bl2, const float* __restrict__ x2t) {
    constexpr int CH = 32;
    __shared__ float g_sh[CH][128];
    __shared__ float p_sh[16][CH];
    __shared__ float d_sh[16][CH];
    __shared__ __align__(16) _Float16 l_sh[16][128];
    __shared__ __align__(16) _Float16 w_sh[128][128];   // [N][K] transposed

    int t = threadIdx.x;
    int b = blockIdx.x >> 5;
    int n0 = (blockIdx.x & 31) * 16;
    _Float16* wdst = &w_sh[0][0];
    for (int idx = t; idx < 128 * 128; idx += 128) wdst[idx] = wl2_t[idx];

    int h = t;  // 0..127
    float w2 = We2[h], b2 = be2[h];
    float acc[16];
#pragma unroll
    for (int r = 0; r < 16; ++r) acc[r] = 0.f;

    for (int c0 = 0; c0 < Cc; c0 += CH) {
        int cn = (Cc - c0 < CH) ? (Cc - c0) : CH;
        __syncthreads();
        for (int idx = t; idx < cn * 128; idx += 128) {
            int cc = idx >> 7, hh = idx & 127;
            g_sh[cc][hh] = g_in[((size_t)b * Nn + c0 + cc) * H2 + hh];
        }
        for (int idx = t; idx < 16 * cn; idx += 128) {
            int r = idx & 15, cc = idx >> 4;
            size_t o = ((size_t)b * Nn + n0 + r) * Cc + c0 + cc;
            p_sh[r][cc] = p_t[o];
            d_sh[r][cc] = d_t[o];
        }
        if (t < 16 && c0 + CH < Cc) {
            size_t o = ((size_t)b * Nn + n0 + t) * Cc + c0 + CH;
            __builtin_prefetch(&p_t[o], 0, 0);
            __builtin_prefetch(&d_t[o], 0, 0);
        }
        __syncthreads();
        for (int cc = 0; cc < cn; ++cc) {
            float g = g_sh[cc][h];
#pragma unroll
            for (int r = 0; r < 16; ++r) {
                acc[r] = fmaf(p_sh[r][cc] * fast_tanh(fmaf(d_sh[r][cc], w2, b2)), g, acc[r]);
            }
        }
    }
#pragma unroll
    for (int r = 0; r < 16; ++r) l_sh[r][h] = (_Float16)acc[r];
    __syncthreads();

    int wv = t >> 5, lane = t & 31, m = lane & 15, hi = lane >> 4;
    v16h A[4];
#pragma unroll
    for (int kk = 0; kk < 4; ++kk) A[kk] = load_frag(&l_sh[m][0], kk * 32, hi);
    for (int ct = wv * 2; ct < wv * 2 + 2; ++ct) {
        v8f c = {0.f, 0.f, 0.f, 0.f, 0.f, 0.f, 0.f, 0.f};
#pragma unroll
        for (int kk = 0; kk < 4; ++kk)
            c = wmma16(A[kk], load_frag(&w_sh[ct * 16 + m][0], kk * 32, hi), c);
#pragma unroll
        for (int v = 0; v < 8; ++v) {
            int row = v + hi * 8, hc = ct * 16 + m, ng = n0 + row;
            float val = c[v] + bl2[hc] + x2t[((size_t)b * Nn + ng) * H2 + hc];
            g_out[((size_t)b * Nn + ng) * H2 + hc] = val > 0.f ? val : 0.f;
        }
    }
}

// final: out[b] = sum_n (gamma@WQ + bQ) * avail[b,n]
__global__ __launch_bounds__(256)
void k_final(const float* __restrict__ gamma, const float* __restrict__ WQ,
             const float* __restrict__ bQ, const float* __restrict__ avail,
             float* __restrict__ out) {
    __shared__ float wq[128];
    __shared__ float red[256];
    int t = threadIdx.x, b = blockIdx.x;
    if (t < 128) wq[t] = WQ[t];
    __syncthreads();
    float bq = bQ[0];
    float s = 0.f;
    for (int n = t; n < Nn; n += 256) {
        const float* g = gamma + ((size_t)b * Nn + n) * H2;
        float dot = 0.f;
        for (int k = 0; k < 128; ++k) dot = fmaf(g[k], wq[k], dot);
        s += (dot + bq) * avail[(size_t)b * Nn + n];
    }
    red[t] = s;
    __syncthreads();
    for (int st = 128; st > 0; st >>= 1) {
        if (t < st) red[t] += red[t + st];
        __syncthreads();
    }
    if (t == 0) out[b] = red[0];
}

// ---------------- host launcher ----------------
extern "C" void kernel_launch(void* const* d_in, const int* in_sizes, int n_in,
                              void* d_out, int out_size, void* d_ws, size_t ws_size,
                              hipStream_t stream) {
    const float* x_a = (const float*)d_in[0];
    const float* ap = (const float*)d_in[1];
    const float* act = (const float*)d_in[2];
    const float* x_b = (const float*)d_in[3];
    const float* edge = (const float*)d_in[4];
    const float* avail = (const float*)d_in[5];
    // d_in[6] avail_robot: unused by reference
    const float* u_a0 = (const float*)d_in[7];
    const float* u_b0 = (const float*)d_in[8];
    const float* g0 = (const float*)d_in[9];
    const float* W1p = (const float*)d_in[10];  const float* b1p = (const float*)d_in[11];
    const float* W2p = (const float*)d_in[12];  const float* b2p = (const float*)d_in[13];
    const float* Wx1a = (const float*)d_in[14]; const float* bx1a = (const float*)d_in[15];
    const float* We1a = (const float*)d_in[16]; const float* be1a = (const float*)d_in[17];
    const float* Wl1a = (const float*)d_in[18]; const float* bl1a = (const float*)d_in[19];
    const float* Wx1b = (const float*)d_in[20]; const float* bx1b = (const float*)d_in[21];
    const float* We1b = (const float*)d_in[22]; const float* be1b = (const float*)d_in[23];
    const float* Wl1b = (const float*)d_in[24]; const float* bl1b = (const float*)d_in[25];
    const float* Wx2 = (const float*)d_in[26];  const float* bx2 = (const float*)d_in[27];
    const float* We2 = (const float*)d_in[28];  const float* be2 = (const float*)d_in[29];
    const float* Wl2 = (const float*)d_in[30];  const float* bl2 = (const float*)d_in[31];
    const float* WQ = (const float*)d_in[32];   const float* bQ = (const float*)d_in[33];

    char* ws = (char*)d_ws;
    size_t off = 0;
    auto take = [&](size_t bytes) -> char* {
        char* p = ws + off;
        off = (off + bytes + 255) & ~(size_t)255;
        return p;
    };
    const size_t BNC = (size_t)Bb * Nn * Cc;
    float* p_t = (float*)take(BNC * 4);
    float* d_t = (float*)take(BNC * 4);
    float* h2buf = (float*)take(BNC * 4);
    float* xa_t = (float*)take((size_t)Bb * Nn * Hh * 4);
    float* xb_t = (float*)take((size_t)Bb * Nn * Hh * 4);
    float* x2t = (float*)take((size_t)Bb * Nn * H2 * 4);
    float* ua[2], *ub[2], *gm[2];
    for (int i = 0; i < 2; ++i) {
        ua[i] = (float*)take((size_t)Bb * Nn * Hh * 4);
        ub[i] = (float*)take((size_t)Bb * Nn * Hh * 4);
        gm[i] = (float*)take((size_t)Bb * Nn * H2 * 4);
    }
    _Float16* wl1a_t = (_Float16*)take(64 * 64 * 2);
    _Float16* wl1b_t = (_Float16*)take(64 * 64 * 2);
    _Float16* wx2_t = (_Float16*)take(128 * 128 * 2);
    _Float16* wl2_t = (_Float16*)take(128 * 128 * 2);

    // weight conversions to f16, stored transposed [N][K] for contiguous B frags
    k_cvt_h_t<<<(4096 + 255) / 256, 256, 0, stream>>>(Wl1a, wl1a_t, 64, 64);
    k_cvt_h_t<<<(4096 + 255) / 256, 256, 0, stream>>>(Wl1b, wl1b_t, 64, 64);
    k_cvt_h_t<<<(16384 + 255) / 256, 256, 0, stream>>>(Wx2, wx2_t, 128, 128);
    k_cvt_h_t<<<(16384 + 255) / 256, 256, 0, stream>>>(Wl2, wl2_t, 128, 128);

    k_xa_terms<<<(Bb * Nn + 127) / 128, 128, 0, stream>>>(
        x_a, ap, act, x_b, Wx1a, bx1a, Wx1b, bx1b, xa_t, xb_t);

    hipMemcpyAsync(ua[0], u_a0, (size_t)Bb * Nn * Hh * 4, hipMemcpyDeviceToDevice, stream);
    hipMemcpyAsync(ub[0], u_b0, (size_t)Bb * Nn * Hh * 4, hipMemcpyDeviceToDevice, stream);
    hipMemcpyAsync(gm[0], g0, (size_t)Bb * Nn * H2 * 4, hipMemcpyDeviceToDevice, stream);

    long tot = (long)Bb * Cc * Nn;
    k_presence<<<(int)((tot + 255) / 256), 256, 0, stream>>>(
        edge, W1p, b1p, W2p, b2p, h2buf, d_t);
    k_softmax<<<Bb * Cc, 256, 0, stream>>>(h2buf, avail, p_t);

    for (int i = 0; i < T1; ++i) {
        int s = i & 1, d = s ^ 1;
        k_iter1<<<Bb * 32, 128, 0, stream>>>(
            p_t, d_t, ua[s], ub[s], ua[d], ub[d], wl1a_t, wl1b_t,
            We1a, be1a, We1b, be1b, bl1a, bl1b, xa_t, xb_t);
    }
    // T1=5 odd -> final in buffer 1
    k_x2term<<<Bb * 32, 128, 0, stream>>>(ua[1], ub[1], wx2_t, bx2, x2t);

    for (int i = 0; i < T2; ++i) {
        int s = i & 1, d = s ^ 1;
        k_iter2<<<Bb * 32, 128, 0, stream>>>(
            p_t, d_t, gm[s], gm[d], wl2_t, We2, be2, bl2, x2t);
    }
    k_final<<<Bb, 256, 0, stream>>>(gm[1], WQ, bQ, avail, (float*)d_out);
}
